// VACNN_12799002542377
// MI455X (gfx1250) — matrix operand order repaired and checked by reading, not compiled, not run on hardware
//
#include <hip/hip_runtime.h>
#include <hip/hip_bf16.h>
#include <stdint.h>

// ---------------------------------------------------------------------------
// Types for CDNA5 WMMA (wave32): A/B = 16 bf16 (8 VGPRs), C/D = 8 f32
// ---------------------------------------------------------------------------
typedef __attribute__((ext_vector_type(16))) __bf16          v16bf;
typedef __attribute__((ext_vector_type(8)))  float           v8f;
typedef __attribute__((ext_vector_type(8)))  unsigned short  v8us;
typedef __attribute__((ext_vector_type(16))) unsigned short  v16us;

__device__ __forceinline__ unsigned short f2bf(float f) {
    unsigned int u = __float_as_uint(f);
    unsigned int r = u + 0x7FFFu + ((u >> 16) & 1u);   // round-to-nearest-even
    return (unsigned short)(r >> 16);
}

// Build a v16bf fragment from two 16-byte LDS loads (no memory round-trip).
__device__ __forceinline__ v16bf make_frag(v8us lo, v8us hi) {
    v16us c = __builtin_shufflevector(lo, hi, 0, 1, 2, 3, 4, 5, 6, 7,
                                              8, 9, 10, 11, 12, 13, 14, 15);
    return __builtin_bit_cast(v16bf, c);
}

// A layout (16-bit 16x32): lanes 0-15 hold K {0..7,16..23}; lanes 16-31 {8..15,24..31}
__device__ __forceinline__ v16bf load_fragA(const unsigned short* lds, int row, int hiHalf) {
    const unsigned short* p = lds + row * 32 + (hiHalf ? 8 : 0);
    v8us lo = *(const v8us*)p;
    v8us hi = *(const v8us*)(p + 16);
    return make_frag(lo, hi);
}

// B layout (16-bit 32x16): lanes 0-15 hold K {0..7,16..23}; lanes 16-31 {8..15,24..31}
__device__ __forceinline__ v16bf load_fragB(const unsigned short* lds, int row, int hiHalf) {
    const unsigned short* p = lds + row * 32 + (hiHalf ? 8 : 0);
    v8us lo = *(const v8us*)p;
    v8us hi = *(const v8us*)(p + 16);
    return make_frag(lo, hi);
}

__device__ __forceinline__ v8f wmma_bf16(v16bf a, v16bf b, v8f c) {
    return __builtin_amdgcn_wmma_f32_16x16x32_bf16(false, a, false, b,
                                                   (short)0, c, false, false);
}

// Epilogue for one 16x16 D tile: (+bias) * gamma + beta, ReLU, scatter NCHW.
// D layout: VGPR r, lanes 0-15 -> M=r, N=lane; lanes 16-31 -> M=8+r, N=lane-16
__device__ __forceinline__ void store_tile(v8f acc, int oc0, int pos, int hiHalf,
                                           int OC, int OH, int OW, int OHW, int numPos,
                                           const float* __restrict__ bias,
                                           const float* __restrict__ gamma,
                                           const float* __restrict__ beta,
                                           float* __restrict__ out) {
    if (pos >= numPos) return;
    int n = pos / OHW; int rem = pos - n * OHW;
    int oy = rem / OW; int ox = rem - oy * OW;
    #pragma unroll
    for (int r = 0; r < 8; ++r) {
        int oc = oc0 + r + (hiHalf ? 8 : 0);
        if (oc >= OC) continue;
        float a = acc[r];
        if (bias) a += bias[oc];
        a = a * gamma[oc] + beta[oc];
        out[(((size_t)n * OC + oc) * OH + oy) * OW + ox] = a > 0.0f ? a : 0.0f;
    }
}

// ---------------------------------------------------------------------------
// Implicit-GEMM conv: out = relu(((W (*) in) [+ bias]) * gamma + beta)
// GEMM: D[OC][pos] = sum_K W[OC][K] * im2col[K][pos],  K = IC*KH*KW
// KH/KW/STRIDE are compile-time so all K-decode divisions constant-fold.
// Block: 256 threads (8 waves). Block tile: 128 OC x 128 positions.
// Wave tile: 32 OC x 64 positions -> 2x4 = 8 WMMA accumulators (pure VGPR).
// LDS per K-step(=32): 32x128 bf16 weights + 32x128 bf16 im2col (16 KB).
// ---------------------------------------------------------------------------
template <int KH, int KW, int STRIDE>
__global__ __launch_bounds__(256)
void conv_wmma_bf16(const float* __restrict__ in, const float* __restrict__ w,
                    const float* __restrict__ bias, const float* __restrict__ gamma,
                    const float* __restrict__ beta, float* __restrict__ out,
                    int N, int IC, int IH, int IW, int OC, int OH, int OW, int pad)
{
    __shared__ __align__(16) unsigned short lds_w[128 * 32]; // [ocL][k]
    __shared__ __align__(16) unsigned short lds_x[128 * 32]; // [posL][k]

    constexpr int KHW = KH * KW;

    const int tid    = threadIdx.x;
    const int wave   = tid >> 5;
    const int lane   = tid & 31;
    const int laneLo = lane & 15;
    const int hiHalf = (lane >= 16) ? 1 : 0;

    const int numPos = N * OH * OW;
    const int Ktot   = IC * KHW;
    const int OHW    = OH * OW;

    const int posBlock = blockIdx.x * 128;
    const int ocBlock  = blockIdx.y * 128;

    const int waveOcOff  = (wave >> 1) * 32;  // 0,32,64,96
    const int wavePosOff = (wave & 1) * 64;   // 0,64

    v8f a00 = {0.f,0.f,0.f,0.f,0.f,0.f,0.f,0.f};
    v8f a01 = a00, a02 = a00, a03 = a00;
    v8f a10 = a00, a11 = a00, a12 = a00, a13 = a00;

    // ---- staging roles (all loop-invariant; hoisted out of the K-loop) ----
    const int rowL  = tid >> 1;          // 0..127: oc row / pos row this thread stages
    const int kHalf = (tid & 1) * 16;    // 0 or 16: which half of the 32-wide K tile

    // weights: contiguous over K
    const int  ocG     = ocBlock + rowL;
    const bool ocValid = (ocG < OC);
    const float* wrow  = w + (size_t)ocG * Ktot;

    // im2col: this thread always stages the same output position
    const int  posG   = posBlock + rowL;
    const bool pValid = (posG < numPos);
    int pn = 0, oy = 0, ox = 0;
    if (pValid) {
        pn = posG / OHW; int rem = posG - pn * OHW;
        oy = rem / OW;   ox = rem - oy * OW;
    }
    const float* inN = in + (size_t)pn * IC * IH * IW;
    const int iy0 = oy * STRIDE - pad;
    const int ix0 = ox * STRIDE - pad;

    unsigned short* ldsWrow = lds_w + rowL * 32 + kHalf;
    unsigned short* ldsXrow = lds_x + rowL * 32 + kHalf;

    for (int k0 = 0; k0 < Ktot; k0 += 32) {
        const int kBase = k0 + kHalf;

        // ---- stage weight tile (sequential f32 stream; prefetch next step) ----
        if (ocValid && kBase + 31 < Ktot)
            __builtin_prefetch(wrow + kBase + 32, 0, 0);
        #pragma unroll
        for (int i = 0; i < 16; ++i) {
            int kg = kBase + i;
            float v = (ocValid && kg < Ktot) ? wrow[kg] : 0.0f;
            ldsWrow[i] = f2bf(v);
        }

        // ---- stage im2col tile: constant-divisor decode + carry increments ----
        {
            int ic = kBase / KHW; int r = kBase - ic * KHW;   // KHW is constexpr
            int ky = r / KW;      int kx = r - ky * KW;       // KW  is constexpr
            #pragma unroll
            for (int i = 0; i < 16; ++i) {
                float v = 0.0f;
                if (pValid && (kBase + i) < Ktot) {
                    int iy = iy0 + ky;
                    int ix = ix0 + kx;
                    if ((unsigned)iy < (unsigned)IH && (unsigned)ix < (unsigned)IW)
                        v = inN[((size_t)ic * IH + iy) * IW + ix];
                }
                ldsXrow[i] = f2bf(v);
                // advance (ic, ky, kx) by one K element (constant trip logic)
                if (++kx == KW) { kx = 0; if (++ky == KH) { ky = 0; ++ic; } }
            }
        }
        __syncthreads();

        // ---- fragments (ds_load_b128 pairs), all SSA values ----
        v16bf fa0 = load_fragA(lds_w, waveOcOff + laneLo,      hiHalf);
        v16bf fa1 = load_fragA(lds_w, waveOcOff + 16 + laneLo, hiHalf);
        v16bf fb0 = load_fragB(lds_x, wavePosOff + laneLo,      hiHalf);
        v16bf fb1 = load_fragB(lds_x, wavePosOff + 16 + laneLo, hiHalf);
        v16bf fb2 = load_fragB(lds_x, wavePosOff + 32 + laneLo, hiHalf);
        v16bf fb3 = load_fragB(lds_x, wavePosOff + 48 + laneLo, hiHalf);

        // ---- 8 WMMAs per K-step ----
        a00 = wmma_bf16(fa0, fb0, a00);
        a01 = wmma_bf16(fa0, fb1, a01);
        a02 = wmma_bf16(fa0, fb2, a02);
        a03 = wmma_bf16(fa0, fb3, a03);
        a10 = wmma_bf16(fa1, fb0, a10);
        a11 = wmma_bf16(fa1, fb1, a11);
        a12 = wmma_bf16(fa1, fb2, a12);
        a13 = wmma_bf16(fa1, fb3, a13);

        __syncthreads();
    }

    const int oc0 = ocBlock + waveOcOff;
    const int pB  = posBlock + wavePosOff + laneLo;
    store_tile(a00, oc0,      pB,      hiHalf, OC, OH, OW, OHW, numPos, bias, gamma, beta, out);
    store_tile(a01, oc0,      pB + 16, hiHalf, OC, OH, OW, OHW, numPos, bias, gamma, beta, out);
    store_tile(a02, oc0,      pB + 32, hiHalf, OC, OH, OW, OHW, numPos, bias, gamma, beta, out);
    store_tile(a03, oc0,      pB + 48, hiHalf, OC, OH, OW, OHW, numPos, bias, gamma, beta, out);
    store_tile(a10, oc0 + 16, pB,      hiHalf, OC, OH, OW, OHW, numPos, bias, gamma, beta, out);
    store_tile(a11, oc0 + 16, pB + 16, hiHalf, OC, OH, OW, OHW, numPos, bias, gamma, beta, out);
    store_tile(a12, oc0 + 16, pB + 32, hiHalf, OC, OH, OW, OHW, numPos, bias, gamma, beta, out);
    store_tile(a13, oc0 + 16, pB + 48, hiHalf, OC, OH, OW, OHW, numPos, bias, gamma, beta, out);
}

// ---------------------------------------------------------------------------
// Small helper kernels
// ---------------------------------------------------------------------------
__global__ void extract_person(const float* __restrict__ x, float* __restrict__ xp,
                               int total, int m) {
    int i = blockIdx.x * blockDim.x + threadIdx.x;
    if (i < total) xp[i] = x[(size_t)i * 2 + m];
}

__global__ void maxpool2x2(const float* __restrict__ in, float* __restrict__ out,
                           int NC, int IH, int IW) {
    int OH = IH >> 1, OW = IW >> 1;
    int total = NC * OH * OW;
    int i = blockIdx.x * blockDim.x + threadIdx.x;
    if (i >= total) return;
    int nc = i / (OH * OW); int rem = i - nc * (OH * OW);
    int y = rem / OW, xo = rem - y * OW;
    size_t base = ((size_t)nc * IH + 2 * y) * IW + 2 * xo;
    float a = fmaxf(in[base], in[base + 1]);
    float b = fmaxf(in[base + IW], in[base + IW + 1]);
    out[i] = fmaxf(a, b);
}

// one block per output element: out[row][j] = dot(in[row,:], w[j,:]) + b[j]
__global__ void fc_rows(const float* __restrict__ in, const float* __restrict__ w,
                        const float* __restrict__ b, float* __restrict__ out,
                        int K, int J) {
    int row = blockIdx.x / J, j = blockIdx.x % J;
    const float* xr = in + (size_t)row * K;
    const float* wr = w  + (size_t)j   * K;
    float s = 0.0f;
    for (int k = threadIdx.x; k < K; k += 256) s += xr[k] * wr[k];
    __shared__ float red[256];
    red[threadIdx.x] = s; __syncthreads();
    for (int off = 128; off > 0; off >>= 1) {
        if (threadIdx.x < off) red[threadIdx.x] += red[threadIdx.x + off];
        __syncthreads();
    }
    if (threadIdx.x == 0) out[(size_t)row * J + j] = red[0] + b[j];
}

// per-sample rotation matrix R = Rx*Ry*Rz and translation bias
__global__ void rot_params(const float* __restrict__ o6, float* __restrict__ Rb, int N) {
    int n = blockIdx.x * blockDim.x + threadIdx.x;
    if (n >= N) return;
    const float PI = 3.14159265358979f;
    const float MINV = -3.602826f;
    float a0 = o6[n*6+0]*PI, a1 = o6[n*6+1]*PI, a2 = o6[n*6+2]*PI;
    float c0 = __cosf(a0), s0 = __sinf(a0);
    float c1 = __cosf(a1), s1 = __sinf(a1);
    float c2 = __cosf(a2), s2 = __sinf(a2);
    // M = Rx*Ry
    float m00 = c1,      m01 = 0.0f, m02 = -s1;
    float m10 = s0*s1,   m11 = c0,   m12 = s0*c1;
    float m20 = c0*s1,   m21 = -s0,  m22 = c0*c1;
    // R = M*Rz ; Rz rows: (c2,s2,0), (-s2,c2,0), (0,0,1)
    float R[9];
    R[0] = m00*c2 - m01*s2; R[1] = m00*s2 + m01*c2; R[2] = m02;
    R[3] = m10*c2 - m11*s2; R[4] = m10*s2 + m11*c2; R[5] = m12;
    R[6] = m20*c2 - m21*s2; R[7] = m20*s2 + m21*c2; R[8] = m22;
    float t0 = MINV - o6[n*6+3], t1 = MINV - o6[n*6+4], t2 = MINV - o6[n*6+5];
    for (int i = 0; i < 9; ++i) Rb[n*12 + i] = R[i];
    for (int i = 0; i < 3; ++i) {
        float shift = R[i*3]*t0 + R[i*3+1]*t1 + R[i*3+2]*t2;
        Rb[n*12 + 9 + i] = 255.0f * (shift - MINV) / 8.812765f;
    }
}

// z[n][i][p] = sum_j R[n][i][j]*xp[n][j][p] + bias[n][i]
__global__ void transform_pts(const float* __restrict__ xp, const float* __restrict__ Rb,
                              float* __restrict__ z, int N, int P) {
    int idx = blockIdx.x * blockDim.x + threadIdx.x;
    if (idx >= N * P) return;
    int n = idx / P, p = idx - n * P;
    const float* rb = Rb + n * 12;
    float x0 = xp[((size_t)n*3 + 0)*P + p];
    float x1 = xp[((size_t)n*3 + 1)*P + p];
    float x2 = xp[((size_t)n*3 + 2)*P + p];
    #pragma unroll
    for (int i = 0; i < 3; ++i)
        z[((size_t)n*3 + i)*P + p] = rb[i*3]*x0 + rb[i*3+1]*x1 + rb[i*3+2]*x2 + rb[9+i];
}

__global__ void pmax_copy_or_max(float* __restrict__ dst, const float* __restrict__ src,
                                 int total, int isMax) {
    int i = blockIdx.x * blockDim.x + threadIdx.x;
    if (i < total) dst[i] = isMax ? fmaxf(dst[i], src[i]) : src[i];
}

// ---------------------------------------------------------------------------
// Orchestration
// ---------------------------------------------------------------------------
extern "C" void kernel_launch(void* const* d_in, const int* in_sizes, int n_in,
                              void* d_out, int out_size, void* d_ws, size_t ws_size,
                              hipStream_t stream)
{
    (void)in_sizes; (void)n_in; (void)out_size; (void)ws_size;

    const float* x   = (const float*)d_in[0];
    const float* w1  = (const float*)d_in[1];
    const float* b1  = (const float*)d_in[2];
    const float* g1  = (const float*)d_in[3];
    const float* be1 = (const float*)d_in[4];
    const float* w2  = (const float*)d_in[5];
    const float* b2  = (const float*)d_in[6];
    const float* g2  = (const float*)d_in[7];
    const float* be2 = (const float*)d_in[8];
    const float* fw  = (const float*)d_in[9];
    const float* fb  = (const float*)d_in[10];
    const float* bw[5]; const float* bg[5]; const float* bb[5];
    for (int i = 0; i < 5; ++i) {
        bw[i] = (const float*)d_in[11 + 3*i];
        bg[i] = (const float*)d_in[12 + 3*i];
        bb[i] = (const float*)d_in[13 + 3*i];
    }
    const float* lw = (const float*)d_in[26];
    const float* lb = (const float*)d_in[27];
    float* outp = (float*)d_out;

    const int N = 32;
    char* ws = (char*)d_ws;
    size_t off = 0;
    auto alloc = [&](size_t bytes) -> float* {
        char* p = ws + off;
        off += (bytes + 255) & ~(size_t)255;
        return (float*)p;
    };
    float* X  = alloc((size_t)N*3*224*224*4);      // person input (persists to transform)
    float* R0 = alloc((size_t)N*128*112*112*4);    // 205.5 MB ring buffer
    float* R1 = alloc((size_t)N*256*56*56*4);      // 102.8 MB ring buffer
    float* R2 = alloc((size_t)N*64*112*112*4);     // 102.8 MB ring buffer
    float* CP = alloc((size_t)N*128*28*28*4);      // pooled features
    float* PM = alloc((size_t)N*2048*7*7*4);       // person-max accumulator
    float* F6 = alloc((size_t)N*6*4);              // fc6 outputs
    float* RB = alloc((size_t)N*12*4);             // rotation (9) + bias (3) per sample

    auto conv5 = [&](const float* in, const float* wgt, const float* bias,
                     const float* g, const float* be, float* out,
                     int IC, int IH, int IW, int OC, int OH, int OW) {
        int numPos = N * OH * OW;
        dim3 grid((numPos + 127) / 128, (OC + 127) / 128);
        conv_wmma_bf16<5, 5, 2><<<grid, 256, 0, stream>>>(in, wgt, bias, g, be, out,
                                                          N, IC, IH, IW, OC, OH, OW, 2);
    };
    auto conv3 = [&](const float* in, const float* wgt,
                     const float* g, const float* be, float* out,
                     int IC, int IH, int IW, int OC, int OH, int OW) {
        int numPos = N * OH * OW;
        dim3 grid((numPos + 127) / 128, (OC + 127) / 128);
        conv_wmma_bf16<3, 3, 2><<<grid, 256, 0, stream>>>(in, wgt, nullptr, g, be, out,
                                                          N, IC, IH, IW, OC, OH, OW, 1);
    };

    for (int m = 0; m < 2; ++m) {
        const int totalX = N * 3 * 224 * 224;
        extract_person<<<(totalX + 255) / 256, 256, 0, stream>>>(x, X, totalX, m);

        // head: conv1 -> conv2 -> maxpool -> fc6
        conv5(X,  w1, b1, g1, be1, R0, 3,   224, 224, 128, 112, 112);
        conv5(R0, w2, b2, g2, be2, R1, 128, 112, 112, 128, 56,  56);
        const int poolTotal = N * 128 * 28 * 28;
        maxpool2x2<<<(poolTotal + 255) / 256, 256, 0, stream>>>(R1, CP, N * 128, 56, 56);
        fc_rows<<<N * 6, 256, 0, stream>>>(CP, fw, fb, F6, 128 * 28 * 28, 6);

        // rotation + point transform of the raw person input
        rot_params<<<1, 32, 0, stream>>>(F6, RB, N);
        const int trTotal = N * 224 * 224;
        transform_pts<<<(trTotal + 255) / 256, 256, 0, stream>>>(X, RB, R2, N, 224 * 224);

        // surrogate backbone: 5x (3x3, stride 2, pad 1) + BN + ReLU
        conv3(R2, bw[0], bg[0], bb[0], R1, 3,    224, 224, 64,   112, 112);
        conv3(R1, bw[1], bg[1], bb[1], R0, 64,   112, 112, 256,  56,  56);
        conv3(R0, bw[2], bg[2], bb[2], R2, 256,  56,  56,  512,  28,  28);
        conv3(R2, bw[3], bg[3], bb[3], R1, 512,  28,  28,  1024, 14,  14);
        conv3(R1, bw[4], bg[4], bb[4], R0, 1024, 14,  14,  2048, 7,   7);

        const int zTotal = N * 2048 * 7 * 7;
        pmax_copy_or_max<<<(zTotal + 255) / 256, 256, 0, stream>>>(PM, R0, zTotal, m);
    }

    // final classifier: [32, 100352] @ lw.T + lb -> [32, 60]
    fc_rows<<<N * 60, 256, 0, stream>>>(PM, lw, lb, outp, 2048 * 7 * 7, 60);
}
